// SelfAttentionPooler_83090437308934
// MI455X (gfx1250) — matrix operand order repaired and checked by reading, not compiled, not run on hardware
//
#include <hip/hip_runtime.h>
#include <hip/hip_bf16.h>
#include <math.h>

typedef __bf16 bf16_t;
typedef __attribute__((ext_vector_type(16))) __bf16 v16bf;
typedef __attribute__((ext_vector_type(8)))  __bf16 v8bf;
typedef __attribute__((ext_vector_type(8)))  float  v8f;

constexpr int Bn = 32, S = 512, D = 1024, H = 3, F = 4096;
constexpr float NEG_INF_F = -1e30f;

// ---------------------------------------------------------------------------
// WMMA fragment loaders (bf16, 16x16x32).
// A-matrix 16x32 (ISA 7.12.2): lanes 0-15 -> row M=lane, K in {0..7, 16..23};
//                              lanes 16-31 -> row M=lane-16, K in {8..15, 24..31}.
// B-matrix 32x16: lane holds column N=lane&15; lanes 0-15 K in {0..7, 16..23},
//                 lanes 16-31 K in {8..15, 24..31}.
// Both loaders read two 16-byte chunks (global_load_b128 / ds_load_b128).
// ---------------------------------------------------------------------------
__device__ inline v16bf frag_a(const bf16_t* base, int ld, int row0, int k0, int lane) {
  int r  = row0 + (lane & 15);
  int kh = (lane >> 4) & 1;
  const bf16_t* p = base + (size_t)r * ld + k0 + kh * 8;
  union { v16bf v; v8bf h[2]; } u;
  u.h[0] = *(const v8bf*)p;
  u.h[1] = *(const v8bf*)(p + 16);
  return u.v;
}

__device__ inline v16bf frag_b(const bf16_t* base, int ld, int n0, int k0, int lane) {
  // memory holds "rows" indexed by n with the k-contraction contiguous
  int n  = n0 + (lane & 15);
  int kh = (lane >> 4) & 1;
  const bf16_t* p = base + (size_t)n * ld + k0 + kh * 8;
  union { v16bf v; v8bf h[2]; } u;
  u.h[0] = *(const v8bf*)p;
  u.h[1] = *(const v8bf*)(p + 16);
  return u.v;
}

__device__ inline v8f wmma_bf16(v16bf a, v16bf b, v8f c) {
  return __builtin_amdgcn_wmma_f32_16x16x32_bf16(false, a, false, b, (short)0, c, false, false);
}

// ---------------------------------------------------------------------------
// bf16 GEMM with pre-transposed B: C[M,N] = act(A[M,K] * Bt[N,K]^T + bias),
// batched over heads. 256 threads (8 waves), workgroup tile 128x64, wave tile
// 32x32 (4 accumulators, 4 WMMA per k-step). A and B staged through
// double-buffered LDS with pure b128 stores/loads; one barrier per k-step.
// LDS row stride 40 elems (80B): 16B-aligned, conflict-free (gcd(20,64)=4).
// mode 0: bias -> bf16;  mode 1: relu -> bf16;  mode 2: tanh -> atomicAdd f32
// ---------------------------------------------------------------------------
constexpr int LDSS = 40;

__global__ __launch_bounds__(256) void gemm_bf16_kernel(
    const bf16_t* __restrict__ A, const bf16_t* __restrict__ Bt,
    const float* __restrict__ bias, void* __restrict__ Cout,
    int M, int N, int K,
    size_t strideA, size_t strideB, size_t strideBias, size_t strideC, int mode)
{
  __shared__ __align__(16) bf16_t As[2][128 * LDSS];
  __shared__ __align__(16) bf16_t Bs[2][64 * LDSS];
  const int h = blockIdx.z;
  A += strideA * h; Bt += strideB * h; bias += strideBias * h;
  const int tid = threadIdx.x, wave = tid >> 5, lane = tid & 31;
  const int m0 = blockIdx.y * 128, n0 = blockIdx.x * 64;
  const int wm = (wave & 3) * 32, wn = (wave >> 2) * 32;

  // cooperative staging: A tile 128x32 (16 elems/thread), B tile 64x32 (8/thread)
  const int arow = tid >> 1, acol = (tid & 1) * 16;
  const int brow = tid >> 2, bcol = (tid & 3) * 8;
  int ar = m0 + arow; if (ar > M - 1) ar = M - 1;   // clamp for small M
  const bf16_t* Ap = A  + (size_t)ar * K + acol;
  const bf16_t* Bp = Bt + (size_t)(n0 + brow) * K + bcol;

  v8f c00 = {}, c01 = {}, c10 = {}, c11 = {};
  v8bf ra0 = *(const v8bf*)(Ap);
  v8bf ra1 = *(const v8bf*)(Ap + 8);
  v8bf rb  = *(const v8bf*)(Bp);
  *(v8bf*)&As[0][arow * LDSS + acol]     = ra0;
  *(v8bf*)&As[0][arow * LDSS + acol + 8] = ra1;
  *(v8bf*)&Bs[0][brow * LDSS + bcol]     = rb;
  __syncthreads();

  for (int k0 = 0; k0 < K; k0 += 32) {
    const int p = (k0 >> 5) & 1;
    const bool more = (k0 + 32) < K;
    if (more) {
      ra0 = *(const v8bf*)(Ap + k0 + 32);
      ra1 = *(const v8bf*)(Ap + k0 + 40);
      rb  = *(const v8bf*)(Bp + k0 + 32);
    }
    v16bf a0 = frag_a(As[p], LDSS, wm,      0, lane);
    v16bf a1 = frag_a(As[p], LDSS, wm + 16, 0, lane);
    v16bf b0 = frag_b(Bs[p], LDSS, wn,      0, lane);
    v16bf b1 = frag_b(Bs[p], LDSS, wn + 16, 0, lane);
    c00 = wmma_bf16(a0, b0, c00);
    c01 = wmma_bf16(a0, b1, c01);
    c10 = wmma_bf16(a1, b0, c10);
    c11 = wmma_bf16(a1, b1, c11);
    if (more) {
      *(v8bf*)&As[p ^ 1][arow * LDSS + acol]     = ra0;
      *(v8bf*)&As[p ^ 1][arow * LDSS + acol + 8] = ra1;
      *(v8bf*)&Bs[p ^ 1][brow * LDSS + bcol]     = rb;
    }
    __syncthreads();
  }

  const int half = (lane >> 4) * 8;
  const int colL = lane & 15;
  bf16_t* Cb = (bf16_t*)Cout + strideC * h;
  float*  Cf = (float*)Cout;
  auto store_acc = [&](const v8f& c, int mrow, int ncol) {
    const int col = n0 + ncol + colL;
    const float bsv = bias[col];
#pragma unroll
    for (int r = 0; r < 8; ++r) {
      int grow = m0 + mrow + r + half;
      if (grow >= M) continue;
      float v = c[r] + bsv;
      if (mode == 0)      Cb[(size_t)grow * N + col] = (bf16_t)v;
      else if (mode == 1) Cb[(size_t)grow * N + col] = (bf16_t)fmaxf(v, 0.f);
      else                atomicAdd(&Cf[(size_t)grow * N + col], tanhf(v));
    }
  };
  store_acc(c00, wm,      wn);
  store_acc(c01, wm,      wn + 16);
  store_acc(c10, wm + 16, wn);
  store_acc(c11, wm + 16, wn + 16);
}

// ---------------------------------------------------------------------------
// Attention column-sum kernel. pooled = (colsum(softmax(QK^T/32))/S) @ V, so
// we only need per-(h,b) column sums of the attention matrix.
// Grid: (S/128, B, H); 8 waves, each wave owns 16 query rows.
// ---------------------------------------------------------------------------
__global__ __launch_bounds__(256) void attn_colsum_kernel(
    const bf16_t* __restrict__ Q, const bf16_t* __restrict__ Kmat,
    const int* __restrict__ amask, float* __restrict__ wbuf)
{
  __shared__ float colsum[S];
  const int b = blockIdx.y, h = blockIdx.z;
  const int tid = threadIdx.x, wave = tid >> 5, lane = tid & 31;
  const size_t hb = (size_t)h * Bn + b;
  const bf16_t* Qp = Q    + hb * S * D;
  const bf16_t* Kp = Kmat + hb * S * D;
  for (int i = tid; i < S; i += 256) colsum[i] = 0.f;
  __syncthreads();

  const int row0 = blockIdx.x * 128 + wave * 16;
  const int colLane = lane & 15;
  const float scale = 0.03125f;  // 1/sqrt(1024)
  float rm[8], rs[8];
#pragma unroll
  for (int r = 0; r < 8; ++r) { rm[r] = NEG_INF_F; rs[r] = 0.f; }

  // pass 1: per-row max and sum of exp
  for (int t0 = 0; t0 < S; t0 += 16) {
    v8f c = {};
    for (int d = 0; d < D; d += 32) {
      c = wmma_bf16(frag_a(Qp, D, row0, d, lane), frag_b(Kp, D, t0, d, lane), c);
    }
    const int mk = amask[b * S + t0 + colLane];
#pragma unroll
    for (int r = 0; r < 8; ++r) {
      float sc = mk ? c[r] * scale : NEG_INF_F;
      float m = sc;
      for (int off = 1; off < 16; off <<= 1) m = fmaxf(m, __shfl_xor(m, off, 32));
      float mn = fmaxf(rm[r], m);
      float ts = __expf(sc - mn);
      for (int off = 1; off < 16; off <<= 1) ts += __shfl_xor(ts, off, 32);
      rs[r] = rs[r] * __expf(rm[r] - mn) + ts;
      rm[r] = mn;
    }
  }
  float inv[8];
#pragma unroll
  for (int r = 0; r < 8; ++r) inv[r] = 1.f / fmaxf(rs[r], 1e-30f);

  // pass 2: column sums of normalized attention
  for (int t0 = 0; t0 < S; t0 += 16) {
    v8f c = {};
    for (int d = 0; d < D; d += 32) {
      c = wmma_bf16(frag_a(Qp, D, row0, d, lane), frag_b(Kp, D, t0, d, lane), c);
    }
    const int mk = amask[b * S + t0 + colLane];
    float cs = 0.f;
#pragma unroll
    for (int r = 0; r < 8; ++r) {
      float sc = mk ? c[r] * scale : NEG_INF_F;
      cs += __expf(sc - rm[r]) * inv[r];
    }
    cs += __shfl_xor(cs, 16, 32);   // fold rows 8..15 half onto same column
    if (lane < 16) atomicAdd(&colsum[t0 + colLane], cs);
  }
  __syncthreads();
  for (int i = tid; i < S; i += 256) atomicAdd(&wbuf[hb * S + i], colsum[i]);
}

// ---------------------------------------------------------------------------
// pooled[h,b,:] = (1/S) * wbuf[h,b,:] . V[h,b,:,:]   -> bf16 for the MLP GEMM
// ---------------------------------------------------------------------------
__global__ __launch_bounds__(256) void pool_kernel(
    const float* __restrict__ wbuf, const bf16_t* __restrict__ V,
    bf16_t* __restrict__ pooled)
{
  __shared__ float w[S];
  const int b = blockIdx.y, h = blockIdx.z, tid = threadIdx.x;
  const size_t hb = (size_t)h * Bn + b;
  const float* wp = wbuf + hb * S;
  for (int i = tid; i < S; i += 256) w[i] = wp[i] * (1.f / (float)S);
  __syncthreads();
  const int d = blockIdx.x * 256 + tid;
  const bf16_t* Vp = V + hb * S * D + d;
  float acc = 0.f;
  for (int t = 0; t < S; ++t) acc += w[t] * (float)Vp[(size_t)t * D];
  pooled[hb * D + d] = (bf16_t)acc;
}

// ---------------------------------------------------------------------------
__global__ void cvt_kernel(const float* __restrict__ in, bf16_t* __restrict__ out, size_t n) {
  size_t i = ((size_t)blockIdx.x * blockDim.x + threadIdx.x) * 8;
#pragma unroll
  for (int j = 0; j < 8; ++j) if (i + j < n) out[i + j] = (bf16_t)in[i + j];
}

// f32 [K,N] -> bf16 [N,K] transposed convert (per head via blockIdx.z)
__global__ __launch_bounds__(256) void cvt_transpose_kernel(
    const float* __restrict__ in, bf16_t* __restrict__ out, int K, int N)
{
  __shared__ float tile[32][33];
  const int h = blockIdx.z;
  in  += (size_t)h * K * N;
  out += (size_t)h * K * N;
  const int n0 = blockIdx.x * 32, k0 = blockIdx.y * 32;
  const int tx = threadIdx.x & 31, ty = threadIdx.x >> 5;
#pragma unroll
  for (int j = ty; j < 32; j += 8)
    tile[j][tx] = in[(size_t)(k0 + j) * N + n0 + tx];
  __syncthreads();
#pragma unroll
  for (int j = ty; j < 32; j += 8)
    out[(size_t)(n0 + j) * K + k0 + tx] = (bf16_t)tile[tx][j];
}

__global__ void zero_kernel(float* __restrict__ p, size_t n) {
  size_t i = (size_t)blockIdx.x * blockDim.x + threadIdx.x;
  if (i < n) p[i] = 0.f;
}

// ---------------------------------------------------------------------------
extern "C" void kernel_launch(void* const* d_in, const int* in_sizes, int n_in,
                              void* d_out, int out_size, void* d_ws, size_t ws_size,
                              hipStream_t stream) {
  const float* X   = (const float*)d_in[0];
  const int*   amk = (const int*)  d_in[1];
  const float* Wq  = (const float*)d_in[2];
  const float* bq  = (const float*)d_in[3];
  const float* Wk  = (const float*)d_in[4];
  const float* bk  = (const float*)d_in[5];
  const float* Wv  = (const float*)d_in[6];
  const float* bv  = (const float*)d_in[7];
  const float* W1  = (const float*)d_in[8];
  const float* b1  = (const float*)d_in[9];
  const float* W2  = (const float*)d_in[10];
  const float* b2  = (const float*)d_in[11];
  float* out = (float*)d_out;

  char* ws = (char*)d_ws;
  size_t off = 0;
  auto take = [&](size_t elems, size_t esz) -> char* {
    char* p = ws + off;
    off += (elems * esz + 255) & ~(size_t)255;
    return p;
  };
  bf16_t* Xb   = (bf16_t*)take((size_t)Bn * S * D, 2);
  bf16_t* Wqt  = (bf16_t*)take((size_t)H * D * D, 2);   // transposed [N,K]
  bf16_t* Wkt  = (bf16_t*)take((size_t)H * D * D, 2);
  bf16_t* Wvt  = (bf16_t*)take((size_t)H * D * D, 2);
  bf16_t* W1t  = (bf16_t*)take((size_t)H * D * F, 2);   // [F,D]
  bf16_t* W2t  = (bf16_t*)take((size_t)H * F * D, 2);   // [D,F]
  bf16_t* Qb   = (bf16_t*)take((size_t)H * Bn * S * D, 2);
  bf16_t* Kb   = (bf16_t*)take((size_t)H * Bn * S * D, 2);
  bf16_t* Vb   = (bf16_t*)take((size_t)H * Bn * S * D, 2);
  float*  wbuf = (float*) take((size_t)H * Bn * S, 4);
  bf16_t* Pb   = (bf16_t*)take((size_t)H * Bn * D, 2);
  bf16_t* H1b  = (bf16_t*)take((size_t)H * Bn * F, 2);

  // zero accumulators (d_out receives atomic head-sum; wbuf receives colsums)
  zero_kernel<<<dim3((out_size + 255) / 256), 256, 0, stream>>>(out, (size_t)out_size);
  zero_kernel<<<dim3(((size_t)H * Bn * S + 255) / 256), 256, 0, stream>>>(wbuf, (size_t)H * Bn * S);

  // activations: plain f32 -> bf16
  {
    size_t n = (size_t)Bn * S * D;
    cvt_kernel<<<dim3((unsigned)((n + 2047) / 2048)), 256, 0, stream>>>(X, Xb, n);
  }
  // weights: f32 [K,N] -> bf16 [N,K] (transposed once; GEMM B-frags become
  // contiguous b128 loads, no LDS scatter)
  cvt_transpose_kernel<<<dim3(D / 32, D / 32, H), 256, 0, stream>>>(Wq, Wqt, D, D);
  cvt_transpose_kernel<<<dim3(D / 32, D / 32, H), 256, 0, stream>>>(Wk, Wkt, D, D);
  cvt_transpose_kernel<<<dim3(D / 32, D / 32, H), 256, 0, stream>>>(Wv, Wvt, D, D);
  cvt_transpose_kernel<<<dim3(F / 32, D / 32, H), 256, 0, stream>>>(W1, W1t, D, F);
  cvt_transpose_kernel<<<dim3(D / 32, F / 32, H), 256, 0, stream>>>(W2, W2t, F, D);

  // QKV projections: [16384 x 1024] * [1024 x 1024] per head
  {
    dim3 grid(D / 64, (Bn * S) / 128, H);
    gemm_bf16_kernel<<<grid, 256, 0, stream>>>(Xb, Wqt, bq, Qb, Bn * S, D, D,
        0, (size_t)D * D, (size_t)D, (size_t)Bn * S * D, 0);
    gemm_bf16_kernel<<<grid, 256, 0, stream>>>(Xb, Wkt, bk, Kb, Bn * S, D, D,
        0, (size_t)D * D, (size_t)D, (size_t)Bn * S * D, 0);
    gemm_bf16_kernel<<<grid, 256, 0, stream>>>(Xb, Wvt, bv, Vb, Bn * S, D, D,
        0, (size_t)D * D, (size_t)D, (size_t)Bn * S * D, 0);
  }

  // attention column sums
  attn_colsum_kernel<<<dim3(S / 128, Bn, H), 256, 0, stream>>>(Qb, Kb, amk, wbuf);

  // pooled = (colsum/S) @ V
  pool_kernel<<<dim3(D / 256, Bn, H), 256, 0, stream>>>(wbuf, Vb, Pb);

  // MLP layer 1: [32 x 1024] * [1024 x 4096], relu -> bf16
  gemm_bf16_kernel<<<dim3(F / 64, 1, H), 256, 0, stream>>>(Pb, W1t, b1, H1b, Bn, F, D,
      (size_t)Bn * D, (size_t)D * F, (size_t)F, (size_t)Bn * F, 1);

  // MLP layer 2: [32 x 4096] * [4096 x 1024], tanh -> atomicAdd head-sum
  gemm_bf16_kernel<<<dim3(D / 64, 1, H), 256, 0, stream>>>(H1b, W2t, b2, out, Bn, D, F,
      (size_t)Bn * F, (size_t)F * D, (size_t)D, 0, 2);
}
